// KANLinear_85495618994499
// MI455X (gfx1250) — compile-verified, matrix operand order checked by reading, not run on hardware
//
#include <hip/hip_runtime.h>
#include <hip/hip_bf16.h>

// ---------------------------------------------------------------------------
// KANLinear forward fused as a single bf16 WMMA GEMM:
//   out[8192,1024] = [silu(x) | rbf_basis(x)] (8192 x 9216)  @  W (9216 x 1024)
// GEMM tiles are DMA'd into LDS by the Tensor Data Mover (TDM), synced with
// TENSORcnt + workgroup barriers. No VGPR staging, K-loop kept un-unrolled so
// accumulators coalesce in place (no v_mov/v_nop trains around v_wmma).
// ---------------------------------------------------------------------------

typedef __attribute__((ext_vector_type(16))) __bf16 v16bf;
typedef __attribute__((ext_vector_type(8)))  float  v8f;
typedef unsigned short u16;

#define IN_F      1024
#define OUT_F     1024
#define GRIDG     8
#define KDIM      (IN_F + IN_F * GRIDG)   // 9216
#define BM        128
#define BN        128
#define BK        32
#define LDK       (BK + 8)                // LDS row pad (80B) to spread banks
#define NKT       (KDIM / BK)             // 288 K-steps

union FragU { v16bf v; uint4 q[2]; };

#if __has_builtin(__builtin_amdgcn_tensor_load_to_lds)
#define USE_TDM 1
typedef __attribute__((ext_vector_type(4))) unsigned int tdm_v4u;
typedef __attribute__((ext_vector_type(8))) int          tdm_v8i;
typedef __attribute__((ext_vector_type(4))) int          tdm_v4i;

// 2-D tile DMA: tile[tile_d1][tile_d0] bf16 elements from a row-major tensor
// (row stride = stride_d0 elements) into LDS at lds_off, inserting a 16-byte
// LDS pad after every 64-byte row (pad_interval=3 -> 16 DW, pad_amount=3 -> 4 DW).
__device__ __forceinline__ void tdm_load_2d(const void* gptr, unsigned lds_off,
                                            unsigned tensor_d0, unsigned tensor_d1,
                                            unsigned tile_d0,   unsigned tile_d1,
                                            unsigned stride_d0)
{
    unsigned long long ga = (unsigned long long)(uintptr_t)gptr;
    tdm_v4u g0;
    g0.x = 1u;                                                  // count=1 (valid user D#)
    g0.y = lds_off;                                             // lds_addr (bytes)
    g0.z = (unsigned)(ga & 0xffffffffu);                        // global_addr[31:0]
    g0.w = (unsigned)((ga >> 32) & 0x01ffffffu) | (2u << 30);   // addr[56:32] | type=2

    tdm_v8i g1;
    g1[0] = (int)((1u << 16)        // data_size = 1 -> 2 bytes
                | (1u << 20)        // pad_enable
                | (3u << 22)        // pad_interval: 16 DWORDs (64B)
                | (3u << 25));      // pad_amount:   4 DWORDs (16B)
    g1[1] = (int)((tensor_d0 & 0xffffu) << 16);                          // dim0[15:0]
    g1[2] = (int)(((tensor_d0 >> 16) & 0xffffu) | ((tensor_d1 & 0xffffu) << 16));
    g1[3] = (int)(((tensor_d1 >> 16) & 0xffffu) | (tile_d0 << 16));      // tile_dim0
    g1[4] = (int)(tile_d1 & 0xffffu);                                    // tile_dim1
    g1[5] = (int)stride_d0;                                              // dim0 stride lo
    g1[6] = 0;
    g1[7] = 0;

    tdm_v4i z4 = {0, 0, 0, 0};
    tdm_v8i z8 = {0, 0, 0, 0, 0, 0, 0, 0};
    __builtin_amdgcn_tensor_load_to_lds(g0, g1, z4, z4, z8, 0);
}
#endif

// ---------------------------------------------------------------------------
// Pass 1: bf16 activation matrix A[m][k]:
//   k in [0,1024)     : silu(x[m][k])
//   k in [1024,9216)  : exp(-(((x[m][i]-grid[g])*invd)^2)), k = 1024 + i*8 + g
// ---------------------------------------------------------------------------
__global__ __launch_bounds__(256) void kan_act_kernel(
    const float* __restrict__ x, __hip_bfloat16* __restrict__ Abig, int total)
{
    int idx = blockIdx.x * 256 + threadIdx.x;
    if (idx >= total) return;
    int m = idx >> 10;
    int i = idx & (IN_F - 1);

    float v   = x[idx];
    float sig = 1.0f / (1.0f + __expf(-v));
    Abig[(size_t)m * KDIM + i] = __float2bfloat16(v * sig);

    const float step = 4.0f / 7.0f;
    const float invd = 1.0f / (4.0f / 7.0f + 1e-6f);
    union { __hip_bfloat16 h[8]; uint4 q; } pk;
#pragma unroll
    for (int g = 0; g < GRIDG; ++g) {
        float t = (v - (-2.0f + step * (float)g)) * invd;
        pk.h[g] = __float2bfloat16(__expf(-t * t));
    }
    *(uint4*)&Abig[(size_t)m * KDIM + IN_F + (size_t)i * GRIDG] = pk.q;
}

// ---------------------------------------------------------------------------
// Pass 2: weights, N-major: WT[o][k] = k<1024 ? base_w[o][k] : spline_w[o][k-1024]
// ---------------------------------------------------------------------------
__global__ __launch_bounds__(256) void kan_wprep_kernel(
    const float* __restrict__ base_w, const float* __restrict__ spline_w,
    __hip_bfloat16* __restrict__ WT)
{
    int o = blockIdx.y;
    int k = blockIdx.x * 256 + threadIdx.x;
    float v = (k < IN_F) ? base_w[(size_t)o * IN_F + k]
                         : spline_w[(size_t)o * (IN_F * GRIDG) + (k - IN_F)];
    WT[(size_t)o * KDIM + k] = __float2bfloat16(v);
}

// ---------------------------------------------------------------------------
// Pass 3: GEMM. Block = 128x128, 256 threads = 8 waves (wave32).
// Wave grid 2x4 (M x N): each wave computes 64x32 = 4x2 tiles of 16x16.
// ---------------------------------------------------------------------------
__global__ __launch_bounds__(256) void kan_gemm_kernel(
    const u16* __restrict__ Abig, const u16* __restrict__ WT,
    const float* __restrict__ base_b, float* __restrict__ out)
{
    __shared__ alignas(16) u16 lA[2][BM][LDK];
    __shared__ alignas(16) u16 lB[2][BN][LDK];

    const int tid  = threadIdx.x;
    const int lane = tid & 31;
    const int wv   = tid >> 5;
    const int wm   = wv >> 2;
    const int wn   = wv & 3;
    const int m0   = blockIdx.y * BM;
    const int n0   = blockIdx.x * BN;
    const int lr   = lane & 15;
    const int c0   = (lane >> 4) * 8;

    v8f acc[4][2];
#pragma unroll
    for (int a = 0; a < 4; ++a)
#pragma unroll
        for (int b = 0; b < 2; ++b)
            acc[a][b] = (v8f){0.f,0.f,0.f,0.f,0.f,0.f,0.f,0.f};

    // Per-lane LDS fragment base pointers (buffer 0); buffer 1 = +stride.
    const u16* fragA0 = &lA[0][wm * 64 + lr][c0];
    const u16* fragB0 = &lB[0][wn * 32 + lr][c0];
    const int  strA   = BM * LDK;   // elements between buffers
    const int  strB   = BN * LDK;

#if USE_TDM
    const unsigned ldsA[2] = { (unsigned)(uintptr_t)&lA[0][0][0],
                               (unsigned)(uintptr_t)&lA[1][0][0] };
    const unsigned ldsB[2] = { (unsigned)(uintptr_t)&lB[0][0][0],
                               (unsigned)(uintptr_t)&lB[1][0][0] };
    auto issue = [&](int buf, int kt) {
        tdm_load_2d(Abig + (size_t)m0 * KDIM + kt, ldsA[buf],
                    KDIM, 8192, BK, BM, KDIM);
        tdm_load_2d(WT + (size_t)n0 * KDIM + kt, ldsB[buf],
                    KDIM, OUT_F, BK, BN, KDIM);
    };
#else
    uint4 ra[2], rb[2];
    auto fetch = [&](int kt) {
#pragma unroll
        for (int j = 0; j < 2; ++j) {
            int c = tid + j * 256, r = c >> 2, c8 = (c & 3) * 8;
            ra[j] = *(const uint4*)(Abig + (size_t)(m0 + r) * KDIM + kt + c8);
            rb[j] = *(const uint4*)(WT   + (size_t)(n0 + r) * KDIM + kt + c8);
        }
    };
    auto stg = [&](int buf) {
#pragma unroll
        for (int j = 0; j < 2; ++j) {
            int c = tid + j * 256, r = c >> 2, c8 = (c & 3) * 8;
            *(uint4*)&lA[buf][r][c8] = ra[j];
            *(uint4*)&lB[buf][r][c8] = rb[j];
        }
    };
#endif

    auto compute = [&](const u16* bufA, const u16* bufB) {
        FragU af[4], bf[2];
#pragma unroll
        for (int ni = 0; ni < 2; ++ni) {
            const u16* p = bufB + (ni * 16) * LDK;
            bf[ni].q[0] = *(const uint4*)(p);
            bf[ni].q[1] = *(const uint4*)(p + 16);
        }
#pragma unroll
        for (int mi = 0; mi < 4; ++mi) {
            const u16* p = bufA + (mi * 16) * LDK;
            af[mi].q[0] = *(const uint4*)(p);
            af[mi].q[1] = *(const uint4*)(p + 16);
        }
#pragma unroll
        for (int mi = 0; mi < 4; ++mi)
#pragma unroll
            for (int ni = 0; ni < 2; ++ni)
                acc[mi][ni] = __builtin_amdgcn_wmma_f32_16x16x32_bf16(
                    false, af[mi].v, false, bf[ni].v,
                    (short)0, acc[mi][ni], false, false);
    };

#if USE_TDM
    if (wv == 0) {
        issue(0, 0);
        __builtin_amdgcn_s_wait_tensorcnt(0);   // tile 0 resident
    }
    __syncthreads();
#pragma clang loop unroll(disable)
    for (int kti = 0; kti < NKT; ++kti) {
        const int odd = kti & 1;
        if (kti + 1 < NKT && wv == 0) issue(odd ^ 1, (kti + 1) * BK);
        compute(fragA0 + odd * strA, fragB0 + odd * strB);
        if (kti + 1 < NKT && wv == 0) __builtin_amdgcn_s_wait_tensorcnt(0);
        __syncthreads();                        // publish next tile / retire cur readers
    }
#else
    fetch(0);
    stg(0);
#pragma clang loop unroll(disable)
    for (int kti = 0; kti < NKT; ++kti) {
        const int odd = kti & 1;
        __syncthreads();
        if (kti + 1 < NKT) fetch((kti + 1) * BK);
        compute(fragA0 + odd * strA, fragB0 + odd * strB);
        if (kti + 1 < NKT) stg(odd ^ 1);
    }
#endif

    // Epilogue: D layout (16x16 f32): row m = r + (lane>=16)*8, col n = lane%16
#pragma unroll
    for (int ni = 0; ni < 2; ++ni) {
        const int n    = n0 + wn * 32 + ni * 16 + lr;
        const float bv = base_b[n];
#pragma unroll
        for (int mi = 0; mi < 4; ++mi) {
            const int mb = m0 + wm * 64 + mi * 16 + (lane >> 4) * 8;
#pragma unroll
            for (int r = 0; r < 8; ++r)
                out[(size_t)(mb + r) * OUT_F + n] = acc[mi][ni][r] + bv;
        }
    }
}

// ---------------------------------------------------------------------------
extern "C" void kernel_launch(void* const* d_in, const int* in_sizes, int n_in,
                              void* d_out, int out_size, void* d_ws, size_t ws_size,
                              hipStream_t stream)
{
    const float* x        = (const float*)d_in[0];   // [4,2048,1024]
    const float* base_w   = (const float*)d_in[1];   // [1024,1024]
    const float* base_b   = (const float*)d_in[2];   // [1024]
    const float* spline_w = (const float*)d_in[3];   // [1024,1024,8]
    float*       out      = (float*)d_out;           // [4,2048,1024]

    const int total = in_sizes[0];                   // 8,388,608
    const int M     = total / IN_F;                  // 8192

    // workspace: Abig [M x 9216] bf16 (151 MB), WT [1024 x 9216] bf16 (19 MB)
    __hip_bfloat16* Abig = (__hip_bfloat16*)d_ws;
    __hip_bfloat16* WT   = Abig + (size_t)M * KDIM;

    kan_act_kernel<<<(total + 255) / 256, 256, 0, stream>>>(x, Abig, total);

    dim3 wg(KDIM / 256, OUT_F);                      // (36, 1024)
    kan_wprep_kernel<<<wg, 256, 0, stream>>>(base_w, spline_w, WT);

    dim3 gg(OUT_F / BN, M / BM);                     // (8, 64)
    kan_gemm_kernel<<<gg, 256, 0, stream>>>(
        (const u16*)Abig, (const u16*)WT, base_b, out);
}
